// PointerDecoder_12094627905810
// MI455X (gfx1250) — compile-verified, matrix operand order checked by reading, not run on hardware
//
#include <hip/hip_runtime.h>
#include <hip/hip_bf16.h>
#include <stdint.h>

// ---------------------------------------------------------------------------
// PointerDecoder for MI455X (gfx1250): persistent WMMA-bf16 recurrent kernel.
// L=512, B=64, D_IN=256, H=512.  All recurrent state + enc_k stay L2-resident.
// enc_k stored bf16 -> dominant per-step traffic halved (32 MB/step from L2).
// ---------------------------------------------------------------------------

typedef __attribute__((ext_vector_type(16))) __bf16 v16bf;
typedef __attribute__((ext_vector_type(8)))  __bf16 v8bf;
typedef __attribute__((ext_vector_type(8)))  float  v8f;

#define L_SEQ  512
#define BATCH  64
#define DIN    256
#define HID    512
#define H3     1536
#define STEPS  511

#define NB     96           // persistent blocks
#define TPB    256          // 8 wave32 per block
#define NWAVES ((NB * TPB) / 32)   // 768 waves

#define WMMA_BF16(a, b, c) \
  __builtin_amdgcn_wmma_f32_16x16x32_bf16(false, (a), false, (b), (short)0, (c), false, false)

// ---------- helpers --------------------------------------------------------

__device__ __forceinline__ float sigmoid_f(float x) {
  return 1.0f / (1.0f + __expf(-x));
}

__device__ __forceinline__ float tanh_f(float x) {
#if __has_builtin(__builtin_amdgcn_tanhf)
  return __builtin_amdgcn_tanhf(x);            // v_tanh_f32 (gfx1250 TRANS op)
#elif __has_builtin(__builtin_amdgcn_tanh_f32)
  return __builtin_amdgcn_tanh_f32(x);
#else
  float e = __expf(-2.0f * fabsf(x));
  float r = (1.0f - e) / (1.0f + e);
  return copysignf(r, x);
#endif
}

// A-fragment (16x32 bf16, MxK): lane l holds row m=l&15; K groups
// {k0..k0+7, k0+16..k0+23} with k0 = kt*32 + (l>>4)*8.
__device__ __forceinline__ v16bf load_a_bf(const __bf16* __restrict__ row, int kt, int lane) {
  int k0 = kt * 32 + ((lane >> 4) << 3);
  v8bf lo = *(const v8bf*)(row + k0);
  v8bf hi = *(const v8bf*)(row + k0 + 16);
  v16bf a;
#pragma unroll
  for (int i = 0; i < 8; ++i) { a[i] = lo[i]; a[i + 8] = hi[i]; }
  return a;
}

__device__ __forceinline__ v16bf load_a_f32(const float* __restrict__ row, int kt, int lane) {
  int k0 = kt * 32 + ((lane >> 4) << 3);
  v16bf a;
#pragma unroll
  for (int i = 0; i < 8; ++i) {
    a[i]     = (__bf16)row[k0 + i];
    a[i + 8] = (__bf16)row[k0 + 16 + i];
  }
  return a;
}

// B fragments pre-swizzled: sw[(kt*NT + nt)*512 + lane*16 + j], one 32B load.
__device__ __forceinline__ v16bf load_b_sw(const __bf16* __restrict__ sw, int kt, int nt,
                                           int NT, int lane) {
  return *(const v16bf*)(sw + (size_t)(kt * NT + nt) * 512 + lane * 16);
}

// ---------- weight swizzle: W[K][N] f32 -> WMMA B-fragment bf16 layout ------

__global__ void swizzle_w_kernel(const float* __restrict__ W, __bf16* __restrict__ sw,
                                 int K, int N) {
  size_t i = (size_t)blockIdx.x * blockDim.x + threadIdx.x;
  if (i >= (size_t)K * N) return;
  int   j    = (int)(i & 15);
  int   lane = (int)((i >> 4) & 31);
  size_t tile = i >> 9;
  int   NT = N >> 4;
  int   nt = (int)(tile % NT);
  int   kt = (int)(tile / NT);
  int   n  = nt * 16 + (lane & 15);
  int   k  = kt * 32 + ((lane >> 4) << 3) + (j < 8 ? j : 16 + (j - 8));
  sw[i] = (__bf16)W[(size_t)k * N + n];
}

// ---------- one-shot GEMMs: C = A(f32,MxK) @ Bsw + bias --------------------

__global__ void __launch_bounds__(TPB)
gemm_pre_kernel(const float* __restrict__ A, const __bf16* __restrict__ Bsw,
                const float* __restrict__ bias, float* __restrict__ Cf,
                __bf16* __restrict__ Cb, int M, int N, int K) {
  int lane = threadIdx.x & 31;
  int wave = (int)((blockIdx.x * TPB + threadIdx.x) >> 5);
  int NT = N >> 4;
  int tiles = (M >> 4) * NT;
  if (wave >= tiles) return;
  int mt = wave / NT, nt = wave % NT;
  const float* row = A + (size_t)(mt * 16 + (lane & 15)) * K;
  int KT = K >> 5;
  v8f acc = {};
  for (int kt = 0; kt < KT; ++kt) {
    v16bf a = load_a_f32(row, kt, lane);
    v16bf b = load_b_sw(Bsw, kt, nt, NT, lane);
    acc = WMMA_BF16(a, b, acc);
  }
  int n = nt * 16 + (lane & 15);
  float bz = bias ? bias[n] : 0.0f;
  int mbase = mt * 16 + ((lane >> 4) << 3);
#pragma unroll
  for (int r = 0; r < 8; ++r) {
    float v = acc[r] + bz;
    if (Cf) Cf[(size_t)(mbase + r) * N + n] = v;
    if (Cb) Cb[(size_t)(mbase + r) * N + n] = (__bf16)v;
  }
}

// ---------- state init -----------------------------------------------------

__global__ void init_state_kernel(const float* __restrict__ eh, float* __restrict__ h,
                                  __bf16* __restrict__ hb,
                                  unsigned long long* __restrict__ keys,
                                  unsigned* __restrict__ bar) {
  int i = blockIdx.x * blockDim.x + threadIdx.x;
  if (i < BATCH * HID) {
    float v = eh[i];
    h[i]  = v;
    hb[i] = (__bf16)v;
  }
  if (i < BATCH) keys[i] = 0xFFFFFFFFull;   // decodes to idx = 0 at step 0
  if (i == 0) { bar[0] = 0u; bar[1] = 0u; }
}

// ---------- grid-wide barrier (persistent kernel) --------------------------

__device__ __forceinline__ void grid_sync(unsigned* bar) {
  __syncthreads();
  if (threadIdx.x == 0) {
    __threadfence();
    volatile unsigned* vcnt = bar;
    volatile unsigned* vgen = bar + 1;
    unsigned g = *vgen;
    unsigned prev = atomicAdd(bar, 1u);
    if (prev == NB - 1) {
      *vcnt = 0u;
      __threadfence();
      *vgen = g + 1u;
    } else {
      while (*vgen == g) __builtin_amdgcn_s_sleep(1);
    }
  }
  __syncthreads();
  __threadfence();   // acquire for all threads in the block
}

// ---------- persistent recurrent decoder -----------------------------------

__global__ void __launch_bounds__(TPB)
decoder_kernel(const __bf16* __restrict__ rnn_in,   // (L*B, H) bf16
               const __bf16* __restrict__ enc_kb,   // (L*B, H) bf16
               const __bf16* __restrict__ Wih_sw, const __bf16* __restrict__ Whh_sw,
               const __bf16* __restrict__ Wq_sw,
               const float* __restrict__ b_ih, const float* __restrict__ b_hh,
               const float* __restrict__ v_att,
               float* __restrict__ h, __bf16* __restrict__ hb,
               float* __restrict__ gi, float* __restrict__ gh, float* __restrict__ q,
               unsigned long long* __restrict__ keys, unsigned* __restrict__ bar,
               float* __restrict__ out) {
  const int lane = threadIdx.x & 31;
  const int tid  = blockIdx.x * TPB + threadIdx.x;
  const int wave = tid >> 5;

  // per-lane contiguous chunk of v_att (matches phase-E load layout)
  float vreg[16];
#pragma unroll
  for (int j = 0; j < 16; ++j) vreg[j] = v_att[lane * 16 + j];

  for (int t = 0; t < STEPS; ++t) {
    // ---- phase B: gi = gather(x) @ W_ih + b_ih ; gh = h @ W_hh + b_hh -----
    for (int job = wave; job < 768; job += NWAVES) {
      bool is_gi = job < 384;
      int  jj = is_gi ? job : job - 384;
      int  mt = jj / 96, nt = jj % 96;
      int  brow = mt * 16 + (lane & 15);          // batch row for this lane
      const __bf16* arow;
      if (is_gi) {
        unsigned long long key = keys[brow];
        unsigned idx = 0xFFFFFFFFu - (unsigned)(key & 0xFFFFFFFFull);
        arow = rnn_in + ((size_t)idx * BATCH + brow) * HID;   // fused gather
      } else {
        arow = hb + (size_t)brow * HID;
      }
      const __bf16* bsw = is_gi ? Wih_sw : Whh_sw;
      v8f acc = {};
#pragma unroll 4
      for (int kt = 0; kt < 16; ++kt) {
        v16bf a = load_a_bf(arow, kt, lane);
        v16bf b = load_b_sw(bsw, kt, nt, 96, lane);
        acc = WMMA_BF16(a, b, acc);
      }
      const float* bias = is_gi ? b_ih : b_hh;
      float* dst = is_gi ? gi : gh;
      int n = nt * 16 + (lane & 15);
      float bz = bias[n];
      int mbase = mt * 16 + ((lane >> 4) << 3);
#pragma unroll
      for (int r = 0; r < 8; ++r) dst[(size_t)(mbase + r) * H3 + n] = acc[r] + bz;
    }
    grid_sync(bar);

    // ---- phase C: elementwise GRU cell; reset argmax keys -----------------
    for (int e = tid; e < BATCH * HID; e += NB * TPB) {
      int b = e >> 9, hh = e & (HID - 1);
      const float* gib = gi + (size_t)b * H3;
      const float* ghb = gh + (size_t)b * H3;
      float r  = sigmoid_f(gib[hh] + ghb[hh]);
      float z  = sigmoid_f(gib[HID + hh] + ghb[HID + hh]);
      float nn = tanh_f(gib[2 * HID + hh] + r * ghb[2 * HID + hh]);
      float hn = (1.0f - z) * nn + z * h[e];
      h[e]  = hn;
      hb[e] = (__bf16)hn;
    }
    if (tid < BATCH) keys[tid] = 0ull;
    grid_sync(bar);

    // ---- phase D: q = h_new @ W_q -----------------------------------------
    for (int job = wave; job < 128; job += NWAVES) {
      int mt = job >> 5, nt = job & 31;
      const __bf16* arow = hb + (size_t)(mt * 16 + (lane & 15)) * HID;
      v8f acc = {};
#pragma unroll 4
      for (int kt = 0; kt < 16; ++kt) {
        v16bf a = load_a_bf(arow, kt, lane);
        v16bf b = load_b_sw(Wq_sw, kt, nt, 32, lane);
        acc = WMMA_BF16(a, b, acc);
      }
      int n = nt * 16 + (lane & 15);
      int mbase = mt * 16 + ((lane >> 4) << 3);
#pragma unroll
      for (int r = 0; r < 8; ++r) q[(size_t)(mbase + r) * HID + n] = acc[r];
    }
    grid_sync(bar);

    // ---- phase E: scores + fused argmax (packed-key atomicMax) ------------
    // Each wave handles one (l,b) row: 1 KB of bf16 enc_k, 32 B per lane.
    for (int p = wave; p < L_SEQ * BATCH; p += NWAVES) {
      int l = p >> 6, b = p & (BATCH - 1);
      const __bf16* ek = enc_kb + ((size_t)l * BATCH + b) * HID + lane * 16;
      const float*  qp = q + (size_t)b * HID + lane * 16;
      int p2 = p + NWAVES;
      if (p2 < L_SEQ * BATCH) {
        const __bf16* ek2 =
            enc_kb + ((size_t)(p2 >> 6) * BATCH + (p2 & (BATCH - 1))) * HID + lane * 16;
        __builtin_prefetch(ek2, 0, 0);            // global_prefetch_b8, 32B/lane
      }
      v8bf e0 = *(const v8bf*)(ek);
      v8bf e1 = *(const v8bf*)(ek + 8);
      float s = 0.0f;
#pragma unroll
      for (int j = 0; j < 8; ++j) {
        s += vreg[j]     * tanh_f(qp[j]     + (float)e0[j]);
        s += vreg[j + 8] * tanh_f(qp[j + 8] + (float)e1[j]);
      }
#pragma unroll
      for (int off = 16; off > 0; off >>= 1) s += __shfl_xor(s, off, 32);
      if (lane == 0) {
        out[((size_t)t * BATCH + b) * L_SEQ + l] = s;
        unsigned u = __float_as_uint(s);
        unsigned ord = (u & 0x80000000u) ? ~u : (u | 0x80000000u);
        unsigned long long key =
            ((unsigned long long)ord << 32) | (unsigned long long)(0xFFFFFFFFu - (unsigned)l);
        atomicMax(keys + b, key);   // ties resolve to smallest l (argmax-first)
      }
    }
    grid_sync(bar);
  }
}

// ---------------------------------------------------------------------------

extern "C" void kernel_launch(void* const* d_in, const int* in_sizes, int n_in,
                              void* d_out, int out_size, void* d_ws, size_t ws_size,
                              hipStream_t stream) {
  (void)in_sizes; (void)n_in; (void)out_size; (void)ws_size;

  const float* inputs  = (const float*)d_in[0];
  // d_in[1] = lengths (unused: all sequences full length)
  const float* enc_out = (const float*)d_in[2];
  const float* enc_hid = (const float*)d_in[3];
  const float* W_lin   = (const float*)d_in[4];
  const float* b_lin   = (const float*)d_in[5];
  const float* W_ih    = (const float*)d_in[6];
  const float* W_hh    = (const float*)d_in[7];
  const float* b_ih    = (const float*)d_in[8];
  const float* b_hh    = (const float*)d_in[9];
  const float* W_q     = (const float*)d_in[10];
  const float* W_k     = (const float*)d_in[11];
  const float* v_att   = (const float*)d_in[12];
  float* out = (float*)d_out;

  char* p = (char*)d_ws;
  auto carve = [&](size_t bytes) -> char* {
    char* r = p;
    p += (bytes + 255) & ~(size_t)255;
    return r;
  };
  __bf16* rnn_in  = (__bf16*)carve((size_t)L_SEQ * BATCH * HID * 2);  // 33.6 MB
  __bf16* enc_kb  = (__bf16*)carve((size_t)L_SEQ * BATCH * HID * 2);  // 33.6 MB
  __bf16* Wlin_sw = (__bf16*)carve((size_t)DIN * HID * 2);
  __bf16* Wk_sw   = (__bf16*)carve((size_t)HID * HID * 2);
  __bf16* Wih_sw  = (__bf16*)carve((size_t)HID * H3 * 2);
  __bf16* Whh_sw  = (__bf16*)carve((size_t)HID * H3 * 2);
  __bf16* Wq_sw   = (__bf16*)carve((size_t)HID * HID * 2);
  float*  gi      = (float*) carve((size_t)BATCH * H3 * 4);
  float*  gh      = (float*) carve((size_t)BATCH * H3 * 4);
  float*  h       = (float*) carve((size_t)BATCH * HID * 4);
  __bf16* hb      = (__bf16*)carve((size_t)BATCH * HID * 2);
  float*  q       = (float*) carve((size_t)BATCH * HID * 4);
  unsigned long long* keys = (unsigned long long*)carve(BATCH * 8);
  unsigned* bar   = (unsigned*)carve(256);

  // 1) swizzle all weights to WMMA B-fragment bf16 layout
  swizzle_w_kernel<<<(DIN * HID + 255) / 256, 256, 0, stream>>>(W_lin, Wlin_sw, DIN, HID);
  swizzle_w_kernel<<<(HID * HID + 255) / 256, 256, 0, stream>>>(W_k, Wk_sw, HID, HID);
  swizzle_w_kernel<<<(HID * H3 + 255) / 256, 256, 0, stream>>>(W_ih, Wih_sw, HID, H3);
  swizzle_w_kernel<<<(HID * H3 + 255) / 256, 256, 0, stream>>>(W_hh, Whh_sw, HID, H3);
  swizzle_w_kernel<<<(HID * HID + 255) / 256, 256, 0, stream>>>(W_q, Wq_sw, HID, HID);

  // 2) one-shot GEMMs: rnn_inputs (bf16 out) and enc_k (bf16 out)
  const int M = L_SEQ * BATCH;                     // 32768
  const int tiles = (M / 16) * (HID / 16);         // 65536 -> 8192 blocks
  gemm_pre_kernel<<<tiles / 8, TPB, 0, stream>>>(inputs, Wlin_sw, b_lin,
                                                 nullptr, rnn_in, M, HID, DIN);
  gemm_pre_kernel<<<tiles / 8, TPB, 0, stream>>>(enc_out, Wk_sw, nullptr,
                                                 nullptr, enc_kb, M, HID, HID);

  // 3) h0 / keys / barrier init
  init_state_kernel<<<(BATCH * HID + 255) / 256, 256, 0, stream>>>(enc_hid, h, hb, keys, bar);

  // 4) persistent recurrent decoder (511 steps, 4 grid barriers per step)
  decoder_kernel<<<NB, TPB, 0, stream>>>(rnn_in, enc_kb, Wih_sw, Whh_sw, Wq_sw,
                                         b_ih, b_hh, v_att, h, hb, gi, gh, q,
                                         keys, bar, out);
}
